// SaddleModel_80822694576459
// MI455X (gfx1250) — matrix operand order repaired and checked by reading, not compiled, not run on hardware
//
#include <hip/hip_runtime.h>
#include <hip/hip_bf16.h>

// ---------------- model constants ----------------
#define B_   256
#define H_   20
#define S_   21          // H + CLS
#define F_   32
#define CC_  24
#define NC_  8
#define D_   256
#define NH_  8
#define DH_  32
#define NL_  2
#define DFF_ 1024

typedef __attribute__((ext_vector_type(8)))  _Float16 v8h;
typedef __attribute__((ext_vector_type(16))) _Float16 v16h;
typedef __attribute__((ext_vector_type(8)))  float    v8f;

__constant__ int c_cat_off[NC_] = {0, 100, 150, 180, 200, 210, 218, 224};

// ---------------- embedding: build f32 residual X[B,S,F,D] ----------------
__global__ __launch_bounds__(256) void embed_kernel(
    const float* __restrict__ xc, const int* __restrict__ xcat,
    const float* __restrict__ wc, const float* __restrict__ bc,
    const float* __restrict__ tab, const float* __restrict__ cls,
    float* __restrict__ X)
{
  int row = blockIdx.x;            // (b*S + s)*F + f
  int d   = threadIdx.x;           // 0..255
  int f   = row % F_;
  int t   = row / F_;
  int s   = t % S_;
  int b   = t / S_;
  float v;
  if (s == H_) {
    v = cls[f * D_ + d];
  } else if (f < CC_) {
    v = xc[(b * H_ + s) * CC_ + f] * wc[f * D_ + d] + bc[f * D_ + d];
  } else {
    int idx = xcat[(b * H_ + s) * NC_ + (f - CC_)] + c_cat_off[f - CC_];
    v = tab[(size_t)idx * D_ + d];
  }
  X[(size_t)row * D_ + d] = v;
}

// ---------------- weight repack: f32 [K,N] -> f16 WMMA B-fragment order ----
// packed[((ni*ksteps + kt)*32 + lane)*16 + h] = W[kt*32 + (lane>>4)*16 + h][ni*16 + (lane&15)]
__global__ __launch_bounds__(256) void repack_kernel(
    const float* __restrict__ W, _Float16* __restrict__ dst, int K, int N)
{
  int i = blockIdx.x * 256 + threadIdx.x;
  if (i >= K * N) return;
  int h    = i & 15;
  int lane = (i >> 4) & 31;
  int t    = i >> 9;
  int ksteps = K >> 5;
  int kt = t % ksteps;
  int ni = t / ksteps;
  int row = kt * 32 + ((lane >> 4) << 4) + h;
  int col = ni * 16 + (lane & 15);
  dst[i] = (_Float16)W[(size_t)row * N + col];
}

// ---------------- f32 -> f16 elementwise ----------------
__global__ __launch_bounds__(256) void f32_to_f16_kernel(
    const float* __restrict__ src, _Float16* __restrict__ dst, size_t n)
{
  size_t i = (size_t)blockIdx.x * 256 + threadIdx.x;
  if (i < n) dst[i] = (_Float16)src[i];
}

// ---------------- layernorm: f32 row -> f16 row (one wave per row) --------
__global__ __launch_bounds__(256) void layernorm_kernel(
    const float* __restrict__ X, const float* __restrict__ g,
    const float* __restrict__ bb, _Float16* __restrict__ Hh, int Mrows)
{
  int row  = (blockIdx.x << 3) + (threadIdx.x >> 5);
  if (row >= Mrows) return;
  int lane = threadIdx.x & 31;
  const float* x = X + (size_t)row * D_ + lane * 8;
  float v[8];
#pragma unroll
  for (int i = 0; i < 8; ++i) v[i] = x[i];
  float s = 0.f;
#pragma unroll
  for (int i = 0; i < 8; ++i) s += v[i];
#pragma unroll
  for (int off = 16; off > 0; off >>= 1) s += __shfl_xor(s, off, 32);
  float mean = s * (1.0f / D_);
  float vs = 0.f;
#pragma unroll
  for (int i = 0; i < 8; ++i) { float dd = v[i] - mean; vs += dd * dd; }
#pragma unroll
  for (int off = 16; off > 0; off >>= 1) vs += __shfl_xor(vs, off, 32);
  float rstd = rsqrtf(vs * (1.0f / D_) + 1e-5f);
  int db = lane * 8;
  _Float16* o = Hh + (size_t)row * D_ + db;
#pragma unroll
  for (int i = 0; i < 8; ++i)
    o[i] = (_Float16)((v[i] - mean) * rstd * g[db + i] + bb[db + i]);
}

// ---------------- WMMA GEMM: C[M,N] = epi(A[M,K] @ Wpacked) ---------------
// Block = 8 waves covering 8 consecutive 64-row tiles of the SAME 16-col
// panel. The packed B panel (ksteps*512 halves, <=32KB) is staged once into
// LDS cooperatively, so the hot loop is ds_load_b128 (B) + global b128 (A)
// + v_wmma, with global_prefetch_b8 hiding the A stream latency.
// ACT: 0=none 1=gelu(tanh) 2=tanh   OUTMODE: 0=f16 store 1=f32 store 2=f32 +=
__device__ __forceinline__ float gelu_tanh(float x) {
  float x3 = x * x * x;
  return 0.5f * x * (1.0f + tanhf(0.7978845608028654f * (x + 0.044715f * x3)));
}

template <bool HAS_BIAS, int ACT, int OUTMODE>
__global__ __launch_bounds__(256) void gemm_wmma_kernel(
    const _Float16* __restrict__ A, const _Float16* __restrict__ Wp,
    const float* __restrict__ bias,
    _Float16* __restrict__ O16, float* __restrict__ O32,
    int M, int N, int K)
{
  __shared__ _Float16 sB[16384];       // up to 32 k-steps * 512 halves = 32 KB
  const int tid    = threadIdx.x;
  const int lane   = tid & 31;
  const int wv     = tid >> 5;         // wave in block: 0..7
  const int ntiles = N >> 4;
  const int ksteps = K >> 5;
  const int ni = blockIdx.x % ntiles;
  const int mi = (blockIdx.x / ntiles) * 8 + wv;

  // ---- cooperative stage of this ni's full B panel into LDS ----
  {
    const uint4* src = (const uint4*)(Wp + (size_t)ni * ksteps * 512);
    uint4* dst = (uint4*)sB;
    const int nchunk = ksteps * 64;    // 8 halves per 16B chunk
    for (int i = tid; i < nchunk; i += 256) dst[i] = src[i];
  }
  __syncthreads();

  if (mi * 64 >= M) return;

  const int am  = lane & 15;           // A row within tile / C column lane
  const int akb = (lane >> 4) << 4;    // A k-base: 0 or 16

  v8f acc[4] = {};
  const _Float16* a0 = A + (size_t)(mi * 64 + am) * K + akb;

  for (int kt = 0; kt < ksteps; ++kt) {
    v16h bf = *(const v16h*)(sB + kt * 512 + lane * 16);
    if (kt + 4 < ksteps)               // prefetch A stream ~4 k-steps ahead
      __builtin_prefetch(a0 + (kt + 4) * 32, 0, 3);
#pragma unroll
    for (int mr = 0; mr < 4; ++mr) {
      const _Float16* ap = a0 + (size_t)(mr * 16) * K + kt * 32;
      v8h lo = *(const v8h*)(ap);
      v8h hi = *(const v8h*)(ap + 8);
      v16h af = __builtin_shufflevector(lo, hi, 0, 1, 2, 3, 4, 5, 6, 7, 8, 9,
                                        10, 11, 12, 13, 14, 15);
      acc[mr] = __builtin_amdgcn_wmma_f32_16x16x32_f16(
          false, af, false, bf, (short)0, acc[mr], false, false);
    }
  }

  const int cn    = ni * 16 + am;      // output column
  const int rbase = (lane >> 4) << 3;  // 0 or 8 row offset within 16-tile
  float bv = 0.f;
  if (HAS_BIAS) bv = bias[cn];
#pragma unroll
  for (int mr = 0; mr < 4; ++mr) {
#pragma unroll
    for (int r = 0; r < 8; ++r) {
      int row = mi * 64 + mr * 16 + rbase + r;
      float v = acc[mr][r] + bv;
      if (ACT == 1) v = gelu_tanh(v);
      else if (ACT == 2) v = tanhf(v);
      size_t idx = (size_t)row * N + cn;
      if (OUTMODE == 0)       O16[idx] = (_Float16)v;
      else if (OUTMODE == 1)  O32[idx] = v;
      else                    O32[idx] += v;
    }
  }
}

// ---------------- sequence attention (over S per (b,f,head)) --------------
__global__ __launch_bounds__(256) void attn_seq_kernel(
    const _Float16* __restrict__ Q, const _Float16* __restrict__ Kd,
    const _Float16* __restrict__ V, const int* __restrict__ amask,
    _Float16* __restrict__ O)
{
  int wave = (blockIdx.x << 3) + (threadIdx.x >> 5);
  int lane = threadIdx.x & 31;
  int h = wave % NH_;
  int t = wave / NH_;
  int f = t % F_;
  int b = t / F_;
  if (b >= B_) return;
  const int cb = h * DH_;
  int qs = (lane < S_) ? lane : 0;

  float q[DH_];
  {
    const _Float16* qp = Q + ((size_t)(b * S_ + qs) * F_ + f) * D_ + cb;
#pragma unroll
    for (int d = 0; d < DH_; ++d) q[d] = (float)qp[d];
  }
  float sc[S_];
#pragma unroll
  for (int k = 0; k < S_; ++k) {
    const _Float16* kp = Kd + ((size_t)(b * S_ + k) * F_ + f) * D_ + cb;
    float acc = 0.f;
#pragma unroll
    for (int d = 0; d < DH_; ++d) acc += q[d] * (float)kp[d];
    bool valid = (k == H_) || (amask[b * H_ + k] != 0);
    sc[k] = valid ? acc * 0.17677669529663687f : -1e9f;
  }
  float mx = -1e30f;
#pragma unroll
  for (int k = 0; k < S_; ++k) mx = fmaxf(mx, sc[k]);
  float sum = 0.f;
#pragma unroll
  for (int k = 0; k < S_; ++k) { sc[k] = __expf(sc[k] - mx); sum += sc[k]; }
  float inv = 1.0f / sum;
  float o[DH_];
#pragma unroll
  for (int d = 0; d < DH_; ++d) o[d] = 0.f;
#pragma unroll
  for (int k = 0; k < S_; ++k) {
    const _Float16* vp = V + ((size_t)(b * S_ + k) * F_ + f) * D_ + cb;
    float a = sc[k] * inv;
#pragma unroll
    for (int d = 0; d < DH_; ++d) o[d] += a * (float)vp[d];
  }
  if (lane < S_) {
    _Float16* op = O + ((size_t)(b * S_ + lane) * F_ + f) * D_ + cb;
#pragma unroll
    for (int d = 0; d < DH_; ++d) op[d] = (_Float16)o[d];
  }
}

// ---------------- feature attention (over F per (b,s,head)) ---------------
__global__ __launch_bounds__(256) void attn_feat_kernel(
    const _Float16* __restrict__ Q, const _Float16* __restrict__ Kd,
    const _Float16* __restrict__ V, _Float16* __restrict__ O)
{
  int wave = (blockIdx.x << 3) + (threadIdx.x >> 5);
  int lane = threadIdx.x & 31;   // query feature index (all 32 active)
  int h = wave % NH_;
  int t = wave / NH_;
  int s = t % S_;
  int b = t / S_;
  if (b >= B_) return;
  const int cb = h * DH_;
  const size_t base = (size_t)(b * S_ + s) * F_;

  float q[DH_];
  {
    const _Float16* qp = Q + (base + lane) * D_ + cb;
#pragma unroll
    for (int d = 0; d < DH_; ++d) q[d] = (float)qp[d];
  }
  float sc[F_];
#pragma unroll
  for (int k = 0; k < F_; ++k) {
    const _Float16* kp = Kd + (base + k) * D_ + cb;
    float acc = 0.f;
#pragma unroll
    for (int d = 0; d < DH_; ++d) acc += q[d] * (float)kp[d];
    sc[k] = acc * 0.17677669529663687f;
  }
  float mx = -1e30f;
#pragma unroll
  for (int k = 0; k < F_; ++k) mx = fmaxf(mx, sc[k]);
  float sum = 0.f;
#pragma unroll
  for (int k = 0; k < F_; ++k) { sc[k] = __expf(sc[k] - mx); sum += sc[k]; }
  float inv = 1.0f / sum;
  float o[DH_];
#pragma unroll
  for (int d = 0; d < DH_; ++d) o[d] = 0.f;
#pragma unroll
  for (int k = 0; k < F_; ++k) {
    const _Float16* vp = V + (base + k) * D_ + cb;
    float a = sc[k] * inv;
#pragma unroll
    for (int d = 0; d < DH_; ++d) o[d] += a * (float)vp[d];
  }
  _Float16* op = O + (base + lane) * D_ + cb;
#pragma unroll
  for (int d = 0; d < DH_; ++d) op[d] = (_Float16)o[d];
}

// ---------------- attention pooling over features (one block per (b,s)) ---
__global__ __launch_bounds__(256) void pool_kernel(
    const float* __restrict__ X, const _Float16* __restrict__ T,
    const float* __restrict__ pv, float* __restrict__ pooled)
{
  int bs  = blockIdx.x;            // 0..B*S-1
  int tid = threadIdx.x;
  __shared__ float part[F_][8];
  __shared__ float attw[F_];
  int f   = tid >> 3;
  int lig = tid & 7;
  size_t rowb = (size_t)bs * F_;
  {
    const _Float16* tp = T + (rowb + f) * D_ + lig * 32;
    const float*    vp = pv + lig * 32;
    float acc = 0.f;
#pragma unroll
    for (int d = 0; d < 32; ++d) acc += (float)tp[d] * vp[d];
    part[f][lig] = acc;
  }
  __syncthreads();
  if (tid < F_) {
    float sm = 0.f;
#pragma unroll
    for (int j = 0; j < 8; ++j) sm += part[tid][j];
    attw[tid] = sm;
  }
  __syncthreads();
  if (tid == 0) {
    float mx = -1e30f;
    for (int k = 0; k < F_; ++k) mx = fmaxf(mx, attw[k]);
    float su = 0.f;
    for (int k = 0; k < F_; ++k) { attw[k] = __expf(attw[k] - mx); su += attw[k]; }
    float iv = 1.0f / su;
    for (int k = 0; k < F_; ++k) attw[k] *= iv;
  }
  __syncthreads();
  float o = 0.f;
  for (int k = 0; k < F_; ++k) o += attw[k] * X[(rowb + k) * D_ + tid];
  pooled[(size_t)bs * D_ + tid] = o;
}

// ---------------- build combined [B*H, 2D] f16 ----------------------------
__global__ __launch_bounds__(256) void combined_kernel(
    const float* __restrict__ pooled, _Float16* __restrict__ C)
{
  int bh = blockIdx.x;             // b*H + h
  int b  = bh / H_;
  int hh = bh % H_;
  int d  = threadIdx.x;
  C[(size_t)bh * (2 * D_) + d]       = (_Float16)pooled[((size_t)b * S_ + hh) * D_ + d];
  C[(size_t)bh * (2 * D_) + D_ + d]  = (_Float16)pooled[((size_t)b * S_ + H_) * D_ + d];
}

// ---------------- zero padded horses --------------------------------------
__global__ __launch_bounds__(256) void mask_kernel(
    float* __restrict__ out, const int* __restrict__ amask)
{
  int bh = blockIdx.x;
  if (amask[bh] == 0) out[(size_t)bh * D_ + threadIdx.x] = 0.f;
}

// ===========================================================================
extern "C" void kernel_launch(void* const* d_in, const int* in_sizes, int n_in,
                              void* d_out, int out_size, void* d_ws, size_t ws_size,
                              hipStream_t stream) {
  (void)in_sizes; (void)n_in; (void)out_size; (void)ws_size;

  // inputs in setup_inputs() dict order (params flattened in insertion order)
  const float* x_cont   = (const float*)d_in[0];
  const int*   x_cat    = (const int*)  d_in[1];
  const int*   amask    = (const int*)  d_in[2];
  const float* w_cont   = (const float*)d_in[3];
  const float* b_cont   = (const float*)d_in[4];
  const float* cat_tab  = (const float*)d_in[5];
  const float* race_cls = (const float*)d_in[6];
  const float* ln1_g    = (const float*)d_in[7];
  const float* ln1_b    = (const float*)d_in[8];
  const float* wq1      = (const float*)d_in[9];
  const float* wk1      = (const float*)d_in[10];
  const float* wv1      = (const float*)d_in[11];
  const float* wo1      = (const float*)d_in[12];
  const float* ln2_g    = (const float*)d_in[13];
  const float* ln2_b    = (const float*)d_in[14];
  const float* wq2      = (const float*)d_in[15];
  const float* wk2      = (const float*)d_in[16];
  const float* wv2      = (const float*)d_in[17];
  const float* wo2      = (const float*)d_in[18];
  const float* ln3_g    = (const float*)d_in[19];
  const float* ln3_b    = (const float*)d_in[20];
  const float* w_ff1    = (const float*)d_in[21];
  const float* b_ff1    = (const float*)d_in[22];
  const float* w_ff2    = (const float*)d_in[23];
  const float* b_ff2    = (const float*)d_in[24];
  const float* pool_w   = (const float*)d_in[25];
  const float* pool_v   = (const float*)d_in[26];
  const float* w_proj   = (const float*)d_in[27];
  const float* b_proj   = (const float*)d_in[28];

  const size_t M0 = (size_t)B_ * S_ * F_;    // 172032 token rows

  // workspace layout (bytes):
  // [0, M0*1024)                         : X residual f32
  // [M0*1024, +M0*512)                   : H16 (LN output / f16 copy of X)
  // [+..., +4*M0*512)                    : Q16,K16,V16,O16 (aliased: FF16/T16/pooled/C16)
  // [+..., ~4.6 MB)                      : packed f16 weights
  char* ws = (char*)d_ws;
  float*    X   = (float*)ws;
  _Float16* H16 = (_Float16*)(ws + M0 * 1024);
  _Float16* Q16 = (_Float16*)(ws + M0 * 1024 + M0 * 512);
  _Float16* K16 = Q16 + M0 * D_;
  _Float16* V16 = K16 + M0 * D_;
  _Float16* O16 = V16 + M0 * D_;
  _Float16* FF16 = Q16;                 // [M0, DFF] spans Q..O region
  _Float16* T16  = Q16;                 // pooling tanh output (after layers)
  float*    pooled = (float*)K16;       // [B*S, D] f32 (after layers)
  _Float16* C16  = V16;                 // [B*H, 2D] combined
  _Float16* PW   = O16 + M0 * D_;       // packed weights

  // ---- repack all weights into WMMA fragment order (deterministic) ----
  size_t pwOff = 0;
  auto pack = [&](const float* W, int K, int N) -> const _Float16* {
    _Float16* dst = PW + pwOff;
    int total = K * N;
    repack_kernel<<<(total + 255) / 256, 256, 0, stream>>>(W, dst, K, N);
    pwOff += (size_t)total;
    return dst;
  };
  const _Float16 *pq1[NL_], *pk1[NL_], *pv1[NL_], *po1[NL_];
  const _Float16 *pq2[NL_], *pk2[NL_], *pv2[NL_], *po2[NL_];
  const _Float16 *pff1[NL_], *pff2[NL_];
  for (int l = 0; l < NL_; ++l) {
    pq1[l]  = pack(wq1 + (size_t)l * D_ * D_,  D_,  D_);
    pk1[l]  = pack(wk1 + (size_t)l * D_ * D_,  D_,  D_);
    pv1[l]  = pack(wv1 + (size_t)l * D_ * D_,  D_,  D_);
    po1[l]  = pack(wo1 + (size_t)l * D_ * D_,  D_,  D_);
    pq2[l]  = pack(wq2 + (size_t)l * D_ * D_,  D_,  D_);
    pk2[l]  = pack(wk2 + (size_t)l * D_ * D_,  D_,  D_);
    pv2[l]  = pack(wv2 + (size_t)l * D_ * D_,  D_,  D_);
    po2[l]  = pack(wo2 + (size_t)l * D_ * D_,  D_,  D_);
    pff1[l] = pack(w_ff1 + (size_t)l * D_ * DFF_, D_,   DFF_);
    pff2[l] = pack(w_ff2 + (size_t)l * DFF_ * D_, DFF_, D_);
  }
  const _Float16* ppool = pack(pool_w, D_, D_);
  const _Float16* pproj = pack(w_proj, 2 * D_, D_);

  // ---- embedding ----
  embed_kernel<<<(int)M0, 256, 0, stream>>>(x_cont, x_cat, w_cont, b_cont,
                                            cat_tab, race_cls, X);

  const int M0i = (int)M0;
  const int lnBlocks = M0i / 8;
  // block = 8 waves x 64 rows = 512 rows, one 16-col panel
  auto gemmBlocks = [](int M, int N) { return (M / 512) * (N / 16); };

  for (int l = 0; l < NL_; ++l) {
    // -- sequence attention sublayer --
    layernorm_kernel<<<lnBlocks, 256, 0, stream>>>(X, ln1_g + l * D_, ln1_b + l * D_, H16, M0i);
    gemm_wmma_kernel<false,0,0><<<gemmBlocks(M0i, D_), 256, 0, stream>>>(H16, pq1[l], nullptr, Q16, nullptr, M0i, D_, D_);
    gemm_wmma_kernel<false,0,0><<<gemmBlocks(M0i, D_), 256, 0, stream>>>(H16, pk1[l], nullptr, K16, nullptr, M0i, D_, D_);
    gemm_wmma_kernel<false,0,0><<<gemmBlocks(M0i, D_), 256, 0, stream>>>(H16, pv1[l], nullptr, V16, nullptr, M0i, D_, D_);
    attn_seq_kernel<<<(B_ * F_ * NH_) / 8, 256, 0, stream>>>(Q16, K16, V16, amask, O16);
    gemm_wmma_kernel<false,0,2><<<gemmBlocks(M0i, D_), 256, 0, stream>>>(O16, po1[l], nullptr, nullptr, X, M0i, D_, D_);

    // -- feature attention sublayer --
    layernorm_kernel<<<lnBlocks, 256, 0, stream>>>(X, ln2_g + l * D_, ln2_b + l * D_, H16, M0i);
    gemm_wmma_kernel<false,0,0><<<gemmBlocks(M0i, D_), 256, 0, stream>>>(H16, pq2[l], nullptr, Q16, nullptr, M0i, D_, D_);
    gemm_wmma_kernel<false,0,0><<<gemmBlocks(M0i, D_), 256, 0, stream>>>(H16, pk2[l], nullptr, K16, nullptr, M0i, D_, D_);
    gemm_wmma_kernel<false,0,0><<<gemmBlocks(M0i, D_), 256, 0, stream>>>(H16, pv2[l], nullptr, V16, nullptr, M0i, D_, D_);
    attn_feat_kernel<<<(B_ * S_ * NH_) / 8, 256, 0, stream>>>(Q16, K16, V16, O16);
    gemm_wmma_kernel<false,0,2><<<gemmBlocks(M0i, D_), 256, 0, stream>>>(O16, po2[l], nullptr, nullptr, X, M0i, D_, D_);

    // -- FFN sublayer --
    layernorm_kernel<<<lnBlocks, 256, 0, stream>>>(X, ln3_g + l * D_, ln3_b + l * D_, H16, M0i);
    gemm_wmma_kernel<true,1,0><<<gemmBlocks(M0i, DFF_), 256, 0, stream>>>(H16, pff1[l], b_ff1 + l * DFF_, FF16, nullptr, M0i, DFF_, D_);
    gemm_wmma_kernel<true,0,2><<<gemmBlocks(M0i, D_), 256, 0, stream>>>(FF16, pff2[l], b_ff2 + l * D_, nullptr, X, M0i, D_, DFF_);
  }

  // ---- attention pooling: T = tanh(X @ pool_w) via WMMA ----
  f32_to_f16_kernel<<<(int)((M0 * D_ + 255) / 256), 256, 0, stream>>>(X, H16, M0 * D_);
  gemm_wmma_kernel<false,2,0><<<gemmBlocks(M0i, D_), 256, 0, stream>>>(H16, ppool, nullptr, T16, nullptr, M0i, D_, D_);
  pool_kernel<<<B_ * S_, 256, 0, stream>>>(X, T16, pool_v, pooled);

  // ---- combine + final projection -> d_out, then mask ----
  combined_kernel<<<B_ * H_, 256, 0, stream>>>(pooled, C16);
  gemm_wmma_kernel<true,0,1><<<gemmBlocks(B_ * H_, D_), 256, 0, stream>>>(C16, pproj, b_proj, nullptr, (float*)d_out, B_ * H_, D_, 2 * D_);
  mask_kernel<<<B_ * H_, 256, 0, stream>>>((float*)d_out, amask);
}
